// Qwen2DecoderLayer_29497835389353
// MI455X (gfx1250) — matrix operand order repaired and checked by reading, not compiled, not run on hardware
//
#include <hip/hip_runtime.h>
#include <hip/hip_bf16.h>

// Qwen2 decoder layer for gfx1250 (MI455X): f16 WMMA GEMMs + flash attention.
// B=4, S=1024 (T=4096), HID=2048, H=16, KV=4, D=128, INTER=8192.

#define T_TOK 4096
#define HID_  2048
#define NH    16
#define NKV   4
#define HD    128
#define INTER_ 8192
#define BATCH 4
#define SEQ   1024
#define KVD   512          // NKV*HD
#define EPS_  1e-6f

typedef _Float16 half_t;
typedef __attribute__((ext_vector_type(8)))  _Float16 v8h;
typedef __attribute__((ext_vector_type(16))) _Float16 v16h;
typedef __attribute__((ext_vector_type(8)))  float    v8f;
typedef __attribute__((ext_vector_type(4)))  int      v4i;

#ifndef __has_builtin
#define __has_builtin(x) 0
#endif
#if __has_builtin(__builtin_amdgcn_global_load_async_to_lds_b128) && \
    __has_builtin(__builtin_amdgcn_s_wait_asynccnt)
// CDNA5 async tensor path: GLOBAL_LOAD_ASYNC_TO_LDS_B128 (ASYNCcnt tracked)
#define ASYNC_COPY16(ldsp, gp)                                        \
  __builtin_amdgcn_global_load_async_to_lds_b128(                     \
      (__attribute__((address_space(1))) v4i*)(gp),                   \
      (__attribute__((address_space(3))) v4i*)(ldsp), 0, 0)
#define ASYNC_WAIT() __builtin_amdgcn_s_wait_asynccnt(0)
#else
#define ASYNC_COPY16(ldsp, gp) (*(v8h*)(ldsp) = *(const v8h*)(gp))
#define ASYNC_WAIT()
#endif

static __device__ __forceinline__ v16h cat8(v8h lo, v8h hi) {
  return __builtin_shufflevector(lo, hi, 0,1,2,3,4,5,6,7,8,9,10,11,12,13,14,15);
}
static __device__ __forceinline__ v8f wmma_f16(v16h a, v16h b, v8f c) {
  // D = A(16x32 f16) x B(32x16 f16) + C(16x16 f32)
  return __builtin_amdgcn_wmma_f32_16x16x32_f16(false, a, false, b, (short)0, c,
                                                false, false);
}

// ---------------------------------------------------------------- utilities
__global__ __launch_bounds__(256) void cvt_f32_to_f16(
    const float* __restrict__ in, half_t* __restrict__ out, long n) {
  long i = (long)blockIdx.x * 256 + threadIdx.x;
  long stride = (long)gridDim.x * 256;
  for (; i < n; i += stride) out[i] = (half_t)in[i];
}

__global__ __launch_bounds__(256) void rmsnorm_f16(
    const float* __restrict__ x, const float* __restrict__ w,
    half_t* __restrict__ o, int C) {
  const int row = blockIdx.x, tid = threadIdx.x;
  const float* xr = x + (size_t)row * C;
  float ss = 0.f;
  for (int i = tid; i < C; i += 256) { float v = xr[i]; ss += v * v; }
  for (int m = 16; m >= 1; m >>= 1) ss += __shfl_xor(ss, m, 32);
  __shared__ float red[8];
  if ((tid & 31) == 0) red[tid >> 5] = ss;
  __syncthreads();
  float tot = 0.f;
  for (int i = 0; i < 8; ++i) tot += red[i];
  const float rs = rsqrtf(tot / (float)C + EPS_);
  half_t* orow = o + (size_t)row * C;
  for (int i = tid; i < C; i += 256) orow[i] = (half_t)(xr[i] * rs * w[i]);
}

// per-(token, head) RMSNorm + RoPE; q additionally folded with 1/sqrt(D)
__global__ __launch_bounds__(128) void qknorm_rope(
    const float* __restrict__ qf, const float* __restrict__ kf,
    const float* __restrict__ cosb, const float* __restrict__ sinb,
    const float* __restrict__ qw, const float* __restrict__ kw,
    half_t* __restrict__ qh, half_t* __restrict__ kh) {
  const int t = blockIdx.x, head = blockIdx.y, d = threadIdx.x;
  const float* src; const float* w; half_t* dst; float scale;
  if (head < NH) {
    src = qf + (size_t)t * HID_ + head * HD;
    dst = qh + (size_t)t * HID_ + head * HD;
    w = qw; scale = 0.08838834764831845f;           // 1/sqrt(128)
  } else {
    const int kvh = head - NH;
    src = kf + (size_t)t * KVD + kvh * HD;
    dst = kh + (size_t)t * KVD + kvh * HD;
    w = kw; scale = 1.f;
  }
  const float v = src[d];
  float ss = v * v;
  for (int m = 16; m >= 1; m >>= 1) ss += __shfl_xor(ss, m, 32);
  __shared__ float red[4];
  __shared__ float buf[HD];
  if ((d & 31) == 0) red[d >> 5] = ss;
  __syncthreads();
  const float tot = red[0] + red[1] + red[2] + red[3];
  const float xn = v * rsqrtf(tot / (float)HD + EPS_) * w[d];
  buf[d] = xn;
  __syncthreads();
  const float other = (d < 64) ? buf[d + 64] : buf[d - 64];
  const float c = cosb[(size_t)t * HD + d], s = sinb[(size_t)t * HD + d];
  const float r = (d < 64) ? (xn * c - other * s) : (xn * c + other * s);
  dst[d] = (half_t)(r * scale);
}

// v f32 [T][KVD] -> vT f16 [B][KV][D][S]
__global__ __launch_bounds__(256) void v_transpose(
    const float* __restrict__ vf, half_t* __restrict__ vT) {
  const int row = blockIdx.x;                 // (b*NKV+kv)*HD + d
  const int d = row & (HD - 1);
  const int bk = row >> 7, kv = bk & (NKV - 1), b = bk >> 2;
  const int s = blockIdx.y * 256 + threadIdx.x;
  vT[(size_t)row * SEQ + s] =
      (half_t)vf[((size_t)(b * SEQ + s)) * KVD + kv * HD + d];
}

// ----------------------------------------------------------------- GEMM
// C[M,N] = A[M,K] (f16 row-major) * B[N,K]^T (f16 row-major = weight layout)
// block tile 128x128, 8 waves of 32x64, K-step 32, double-buffered LDS,
// next slab staged (async-to-LDS when available) while current slab computes.
// EPI: 0=+bias->f32, 1=+resid->f32, 2=silu->f16, 3=aux*acc->f16
template <int EPI>
__global__ __launch_bounds__(256) void gemm_f16(
    const half_t* __restrict__ A, const half_t* __restrict__ B,
    const float* __restrict__ bias, const float* __restrict__ resid,
    const half_t* __restrict__ aux, float* __restrict__ outF,
    half_t* __restrict__ outH, int M, int N, int K) {
  __shared__ __align__(32) half_t As[2][128 * 32];
  __shared__ __align__(32) half_t Bs[2][128 * 32];
  const int tid = threadIdx.x, lane = tid & 31, wave = tid >> 5;
  const int wm = wave & 3, wn = wave >> 2;       // 4 waves over M, 2 over N
  const int lr = lane & 15, hi = lane >> 4;
  const int m0 = blockIdx.y * 128, n0 = blockIdx.x * 128;
  const v8f zero = {0, 0, 0, 0, 0, 0, 0, 0};
  v8f c[2][4];
  for (int mi = 0; mi < 2; ++mi)
    for (int ni = 0; ni < 4; ++ni) c[mi][ni] = zero;

  const int row0 = tid >> 2, c80 = (tid & 3) * 8;            // chunk 0
  const int row1 = (tid + 256) >> 2, c81 = ((tid + 256) & 3) * 8;  // chunk 1

  // prologue: stage slab 0
  ASYNC_COPY16(&As[0][row0 * 32 + c80], A + (size_t)(m0 + row0) * K + c80);
  ASYNC_COPY16(&Bs[0][row0 * 32 + c80], B + (size_t)(n0 + row0) * K + c80);
  ASYNC_COPY16(&As[0][row1 * 32 + c81], A + (size_t)(m0 + row1) * K + c81);
  ASYNC_COPY16(&Bs[0][row1 * 32 + c81], B + (size_t)(n0 + row1) * K + c81);

  const int nsteps = K >> 5;
  for (int s = 0; s < nsteps; ++s) {
    ASYNC_WAIT();
    __syncthreads();
    if (s + 1 < nsteps) {               // stage next slab while computing
      const int k1 = (s + 1) << 5;
      half_t* an = As[(s + 1) & 1];
      half_t* bn = Bs[(s + 1) & 1];
      ASYNC_COPY16(an + row0 * 32 + c80, A + (size_t)(m0 + row0) * K + k1 + c80);
      ASYNC_COPY16(bn + row0 * 32 + c80, B + (size_t)(n0 + row0) * K + k1 + c80);
      ASYNC_COPY16(an + row1 * 32 + c81, A + (size_t)(m0 + row1) * K + k1 + c81);
      ASYNC_COPY16(bn + row1 * 32 + c81, B + (size_t)(n0 + row1) * K + k1 + c81);
    }
    const half_t* Ab = As[s & 1];
    const half_t* Bb = Bs[s & 1];
    v16h a[2], bfr[4];
#pragma unroll
    for (int mi = 0; mi < 2; ++mi) {
      const half_t* p = Ab + (wm * 32 + mi * 16 + lr) * 32;
      a[mi] = cat8(*(const v8h*)(p + hi * 8), *(const v8h*)(p + 16 + hi * 8));
    }
#pragma unroll
    for (int ni = 0; ni < 4; ++ni) {
      const half_t* p = Bb + (wn * 64 + ni * 16 + lr) * 32;
      bfr[ni] = cat8(*(const v8h*)(p + hi * 8), *(const v8h*)(p + 16 + hi * 8));
    }
#pragma unroll
    for (int mi = 0; mi < 2; ++mi)
#pragma unroll
      for (int ni = 0; ni < 4; ++ni)
        c[mi][ni] = wmma_f16(a[mi], bfr[ni], c[mi][ni]);
  }

#pragma unroll
  for (int mi = 0; mi < 2; ++mi)
#pragma unroll
    for (int v = 0; v < 8; ++v) {
      const int gm = m0 + wm * 32 + mi * 16 + v + hi * 8;
#pragma unroll
      for (int ni = 0; ni < 4; ++ni) {
        const int gn = n0 + wn * 64 + ni * 16 + lr;
        const size_t idx = (size_t)gm * N + gn;
        float acc = c[mi][ni][v];
        if constexpr (EPI == 0) {
          if (bias) acc += bias[gn];
          outF[idx] = acc;
        } else if constexpr (EPI == 1) {
          outF[idx] = acc + resid[idx];
        } else if constexpr (EPI == 2) {
          outH[idx] = (half_t)(acc / (1.f + __expf(-acc)));
        } else {
          outH[idx] = (half_t)((float)aux[idx] * acc);
        }
      }
    }
}

// ------------------------------------------------------------- attention
// grid (SEQ/128, B*H); block 256 = 8 waves; wave owns 16 q-rows, steps k by 32.
__global__ __launch_bounds__(256) void flash_attn(
    const half_t* __restrict__ qh,   // [T][H*D] (pre-scaled by 1/sqrt(D))
    const half_t* __restrict__ kh,   // [T][KV*D]
    const half_t* __restrict__ vT,   // [B][KV][D][S]
    half_t* __restrict__ attnH) {    // [T][H*D]
  __shared__ __align__(32) half_t Ks[32 * 128];   // [k-row][d]
  __shared__ __align__(32) half_t Vs[128 * 32];   // [d][k-col]
  __shared__ __align__(32) half_t Ps[8][16 * 32]; // per-wave P transpose
  const int tid = threadIdx.x, lane = tid & 31, wave = tid >> 5;
  const int lr = lane & 15, hi = lane >> 4;
  const int qb = blockIdx.x * 128;
  const int bh = blockIdx.y, b = bh >> 4, h = bh & 15, kv = h >> 2;
  const int q0 = qb + wave * 16;

  const half_t* qrow = qh + ((size_t)(b * SEQ + q0 + lr)) * HID_ + h * HD;
  v16h qa[4];
#pragma unroll
  for (int dc = 0; dc < 4; ++dc)
    qa[dc] = cat8(*(const v8h*)(qrow + dc * 32 + hi * 8),
                  *(const v8h*)(qrow + dc * 32 + 16 + hi * 8));

  const v8f zero = {0, 0, 0, 0, 0, 0, 0, 0};
  v8f o[8];
  for (int i = 0; i < 8; ++i) o[i] = zero;
  float mrow[8], lrow[8];
  for (int v = 0; v < 8; ++v) { mrow[v] = -__builtin_inff(); lrow[v] = 0.f; }

  const half_t* kbase = kh + (size_t)b * SEQ * KVD + kv * HD;
  const half_t* vbase = vT + ((size_t)(b * NKV + kv)) * HD * SEQ;

  for (int kc = 0; kc < qb + 128; kc += 32) {
#pragma unroll
    for (int i = 0; i < 2; ++i) {
      const int ch = tid + 256 * i;
      const int kr = ch >> 4, kc8 = (ch & 15) * 8;
      ASYNC_COPY16(Ks + kr * 128 + kc8,
                   kbase + (size_t)(kc + kr) * KVD + kc8);
      const int vr = ch >> 2, vc8 = (ch & 3) * 8;
      ASYNC_COPY16(Vs + vr * 32 + vc8,
                   vbase + (size_t)vr * SEQ + kc + vc8);
    }
    ASYNC_WAIT();
    __syncthreads();
    if (kc <= q0 + 15) {                         // wave-uniform causal skip
      v8f s0 = zero, s1 = zero;
#pragma unroll
      for (int dc = 0; dc < 4; ++dc) {
        v16h kb0 = cat8(*(const v8h*)(Ks + lr * 128 + dc * 32 + hi * 8), *(const v8h*)(Ks + lr * 128 + dc * 32 + 16 + hi * 8));
        v16h kb1 = cat8(*(const v8h*)(Ks + (16 + lr) * 128 + dc * 32 + hi * 8), *(const v8h*)(Ks + (16 + lr) * 128 + dc * 32 + 16 + hi * 8));
        s0 = wmma_f16(qa[dc], kb0, s0);
        s1 = wmma_f16(qa[dc], kb1, s1);
      }
#pragma unroll
      for (int v = 0; v < 8; ++v) {
        const int row = q0 + v + hi * 8;
        float a0 = (kc + lr > row) ? -__builtin_inff() : s0[v];
        float a1 = (kc + 16 + lr > row) ? -__builtin_inff() : s1[v];
        float mx = fmaxf(a0, a1);
        for (int m = 8; m >= 1; m >>= 1) mx = fmaxf(mx, __shfl_xor(mx, m, 32));
        const float mn = fmaxf(mrow[v], mx);
        const float alpha = __expf(mrow[v] - mn);
        const float p0 = __expf(a0 - mn);
        const float p1 = __expf(a1 - mn);
        float rs = p0 + p1;
        for (int m = 8; m >= 1; m >>= 1) rs += __shfl_xor(rs, m, 32);
        lrow[v] = lrow[v] * alpha + rs;
        mrow[v] = mn;
#pragma unroll
        for (int nt = 0; nt < 8; ++nt) o[nt][v] = o[nt][v] * alpha;
        half_t* pr = &Ps[wave][(v + hi * 8) * 32];
        pr[lr] = (half_t)p0;
        pr[16 + lr] = (half_t)p1;
      }
      __asm__ __volatile__("" ::: "memory");
      __builtin_amdgcn_wave_barrier();           // same-wave LDS RAW ordering
      v16h pa = cat8(*(const v8h*)&Ps[wave][lr * 32 + hi * 8],
                     *(const v8h*)&Ps[wave][lr * 32 + 16 + hi * 8]);
#pragma unroll
      for (int nt = 0; nt < 8; ++nt) {
        v16h vb = cat8(*(const v8h*)(Vs + (nt * 16 + lr) * 32 + hi * 8), *(const v8h*)(Vs + (nt * 16 + lr) * 32 + 16 + hi * 8));
        o[nt] = wmma_f16(pa, vb, o[nt]);
      }
    }
    __syncthreads();
  }
#pragma unroll
  for (int v = 0; v < 8; ++v) {
    const float inv = 1.f / lrow[v];
    const size_t trow = ((size_t)(b * SEQ + q0 + v + hi * 8)) * HID_ + h * HD;
#pragma unroll
    for (int nt = 0; nt < 8; ++nt)
      attnH[trow + nt * 16 + lr] = (half_t)(o[nt][v] * inv);
  }
}

// ------------------------------------------------------------------ host
extern "C" void kernel_launch(void* const* d_in, const int* in_sizes, int n_in,
                              void* d_out, int out_size, void* d_ws,
                              size_t ws_size, hipStream_t stream) {
  (void)in_sizes; (void)n_in; (void)out_size; (void)ws_size;
  const float* x    = (const float*)d_in[0];
  const float* cosb = (const float*)d_in[2];
  const float* sinb = (const float*)d_in[3];
  const float* wq   = (const float*)d_in[4];
  const float* bq   = (const float*)d_in[5];
  const float* wk   = (const float*)d_in[6];
  const float* bk   = (const float*)d_in[7];
  const float* wv   = (const float*)d_in[8];
  const float* bv   = (const float*)d_in[9];
  const float* wo   = (const float*)d_in[10];
  const float* qnw  = (const float*)d_in[11];
  const float* knw  = (const float*)d_in[12];
  const float* ln1  = (const float*)d_in[13];
  const float* ln2  = (const float*)d_in[14];
  const float* wg   = (const float*)d_in[15];
  const float* wu   = (const float*)d_in[16];
  const float* wd   = (const float*)d_in[17];
  float* out = (float*)d_out;

  // workspace arena (~348 MB); sg aliases the dead [h1,qf,kf,vf] block (67 MB)
  char* base = (char*)d_ws;
  size_t off = 0;
  auto alloc = [&](size_t bytes) -> char* {
    char* p = base + off;
    off += (bytes + 255) & ~(size_t)255;
    return p;
  };
  half_t* h1 = (half_t*)alloc((size_t)T_TOK * HID_ * 2);
  float*  qf = (float*) alloc((size_t)T_TOK * HID_ * 4);
  float*  kf = (float*) alloc((size_t)T_TOK * KVD * 4);
  float*  vf = (float*) alloc((size_t)T_TOK * KVD * 4);
  half_t* sg = (half_t*)base;                    // [T][INTER] f16, aliased
  half_t* qh = (half_t*)alloc((size_t)T_TOK * HID_ * 2);
  half_t* kh = (half_t*)alloc((size_t)T_TOK * KVD * 2);
  half_t* vT = (half_t*)alloc((size_t)T_TOK * KVD * 2);
  half_t* ah = (half_t*)alloc((size_t)T_TOK * HID_ * 2);
  float*  x1 = (float*) alloc((size_t)T_TOK * HID_ * 4);
  half_t* h2 = (half_t*)alloc((size_t)T_TOK * HID_ * 2);
  half_t* act= (half_t*)alloc((size_t)T_TOK * INTER_ * 2);
  half_t* wqh= (half_t*)alloc((size_t)HID_ * HID_ * 2);
  half_t* wkh= (half_t*)alloc((size_t)KVD * HID_ * 2);
  half_t* wvh= (half_t*)alloc((size_t)KVD * HID_ * 2);
  half_t* woh= (half_t*)alloc((size_t)HID_ * HID_ * 2);
  half_t* wgh= (half_t*)alloc((size_t)INTER_ * HID_ * 2);
  half_t* wuh= (half_t*)alloc((size_t)INTER_ * HID_ * 2);
  half_t* wdh= (half_t*)alloc((size_t)HID_ * INTER_ * 2);

  cvt_f32_to_f16<<<4096, 256, 0, stream>>>(wq, wqh, (long)HID_ * HID_);
  cvt_f32_to_f16<<<2048, 256, 0, stream>>>(wk, wkh, (long)KVD * HID_);
  cvt_f32_to_f16<<<2048, 256, 0, stream>>>(wv, wvh, (long)KVD * HID_);
  cvt_f32_to_f16<<<4096, 256, 0, stream>>>(wo, woh, (long)HID_ * HID_);
  cvt_f32_to_f16<<<8192, 256, 0, stream>>>(wg, wgh, (long)INTER_ * HID_);
  cvt_f32_to_f16<<<8192, 256, 0, stream>>>(wu, wuh, (long)INTER_ * HID_);
  cvt_f32_to_f16<<<8192, 256, 0, stream>>>(wd, wdh, (long)HID_ * INTER_);

  rmsnorm_f16<<<T_TOK, 256, 0, stream>>>(x, ln1, h1, HID_);

  gemm_f16<0><<<dim3(HID_ / 128, T_TOK / 128), 256, 0, stream>>>(
      h1, wqh, bq, nullptr, nullptr, qf, nullptr, T_TOK, HID_, HID_);
  gemm_f16<0><<<dim3(KVD / 128, T_TOK / 128), 256, 0, stream>>>(
      h1, wkh, bk, nullptr, nullptr, kf, nullptr, T_TOK, KVD, HID_);
  gemm_f16<0><<<dim3(KVD / 128, T_TOK / 128), 256, 0, stream>>>(
      h1, wvh, bv, nullptr, nullptr, vf, nullptr, T_TOK, KVD, HID_);

  qknorm_rope<<<dim3(T_TOK, NH + NKV), 128, 0, stream>>>(
      qf, kf, cosb, sinb, qnw, knw, qh, kh);
  v_transpose<<<dim3(BATCH * NKV * HD, SEQ / 256), 256, 0, stream>>>(vf, vT);

  flash_attn<<<dim3(SEQ / 128, BATCH * NH), 256, 0, stream>>>(qh, kh, vT, ah);

  gemm_f16<1><<<dim3(HID_ / 128, T_TOK / 128), 256, 0, stream>>>(
      ah, woh, nullptr, x, nullptr, x1, nullptr, T_TOK, HID_, HID_);

  rmsnorm_f16<<<T_TOK, 256, 0, stream>>>(x1, ln2, h2, HID_);

  gemm_f16<2><<<dim3(INTER_ / 128, T_TOK / 128), 256, 0, stream>>>(
      h2, wgh, nullptr, nullptr, nullptr, nullptr, sg, T_TOK, INTER_, HID_);
  gemm_f16<3><<<dim3(INTER_ / 128, T_TOK / 128), 256, 0, stream>>>(
      h2, wuh, nullptr, nullptr, sg, nullptr, act, T_TOK, INTER_, HID_);
  gemm_f16<1><<<dim3(HID_ / 128, T_TOK / 128), 256, 0, stream>>>(
      act, wdh, nullptr, x1, nullptr, out, nullptr, T_TOK, HID_, INTER_);
}
